// Decoder_34479997453075
// MI455X (gfx1250) — compile-verified, matrix operand order checked by reading, not run on hardware
//
#include <hip/hip_runtime.h>
#include <cmath>

#define VOCAB 32000
#define EMB   512
#define HID   1024
#define G3    3072      // 3*HID
#define BATCH 32
#define SEQ   64
#define NROWS 2048      // SEQ*BATCH
#define SOS_ID 1
#define EOS_ID 2
#define PAD_ID 0

typedef __attribute__((ext_vector_type(16))) __bf16 v16bf;
typedef __attribute__((ext_vector_type(8)))  float  v8f;

__device__ inline v8f wmma_bf16(v16bf a, v16bf b, v8f c) {
  return __builtin_amdgcn_wmma_f32_16x16x32_bf16(
      /*neg_a=*/false, a, /*neg_b=*/false, b,
      /*c_mod=*/(short)0, c, /*reuse_a=*/false, /*reuse_b=*/false);
}

// --- CDNA5 async global->LDS copy (ASYNCcnt-tracked, bypasses VGPRs) --------
// global_load_async_to_lds_b128: VDST = VGPR holding LDS byte address,
// VADDR = VGPR pair holding 64-bit global address (ISA §15.18.3 op 98).
__device__ inline void async_copy_b128(unsigned lds_addr, const void* gsrc) {
  asm volatile("global_load_async_to_lds_b128 %0, %1, off"
               :: "v"(lds_addr), "v"((unsigned long long)(uintptr_t)gsrc)
               : "memory");
}
__device__ inline void wait_asynccnt0() {
  asm volatile("s_wait_asynccnt 0x0" ::: "memory");
}
__device__ inline unsigned lds_addr_of(const void* p) {
  return (unsigned)(unsigned long long)(uintptr_t)p;  // low 32 bits = LDS offset
}

// ---------------------------------------------------------------------------
// fp32 -> bf16 cast (one-time weight/state shadowing)
// ---------------------------------------------------------------------------
__global__ void cast_f32_bf16(const float* __restrict__ src,
                              __bf16* __restrict__ dst, int n) {
  int i = blockIdx.x * blockDim.x + threadIdx.x;
  if (i < n) dst[i] = (__bf16)src[i];
}

// ---------------------------------------------------------------------------
// gi = emb[x_t] @ w_ih^T + b_ih  for all (t,b) rows at once.  2048 x 3072, K=512
// WG: 256 thr (8 waves), tile 16 rows x 128 cols; A gathered+cvt into LDS.
// ---------------------------------------------------------------------------
__global__ __launch_bounds__(256) void gi_gemm(
    const int* __restrict__ target, const float* __restrict__ emb,
    const __bf16* __restrict__ wihb, const float* __restrict__ b_ih,
    float* __restrict__ gi)
{
  __shared__ __align__(32) __bf16 As[16 * EMB];   // 16 KB
  __shared__ int tok[16];
  const int rowTile = blockIdx.x;      // 0..127
  const int colTile = blockIdx.y;      // 0..23
  const int tid  = threadIdx.x;
  const int wave = tid >> 5;
  const int lane = tid & 31;

  if (tid < 16) {
    int row = rowTile * 16 + tid;      // row = t*BATCH + b
    int t = row >> 5, b = row & 31;
    tok[tid] = (t == 0) ? SOS_ID : target[(t - 1) * BATCH + b];
  }
  __syncthreads();
  for (int idx = tid; idx < 16 * EMB; idx += 256) {
    int r = idx >> 9;                  // /EMB
    int k = idx & (EMB - 1);
    As[idx] = (__bf16)emb[(size_t)tok[r] * EMB + k];
  }
  __syncthreads();

  const int nn = lane & 15, kh = lane >> 4;
  const int colbase = colTile * 128 + wave * 16;
  const int col = colbase + nn;
  v8f acc = {};
  const __bf16* bptr = wihb + (size_t)col * EMB + kh * 16;
  const __bf16* aptr = As + nn * EMB + kh * 16;   // A row m = lane&15
#pragma unroll 4
  for (int ks = 0; ks < EMB / 32; ++ks) {
    v16bf a = *(const v16bf*)(aptr + ks * 32);
    v16bf b = *(const v16bf*)(bptr + ks * 32);
    acc = wmma_bf16(a, b, acc);
  }
  float bias = b_ih[col];
  int rowb = rowTile * 16 + kh * 8;    // VGPR i -> row i (+8 for hi lanes)
#pragma unroll
  for (int i = 0; i < 8; ++i)
    gi[(size_t)(rowb + i) * G3 + col] = acc[i] + bias;
}

// ---------------------------------------------------------------------------
// One GRU recurrence step (serial; launched SEQ times).
// WG: 128 thr (4 waves); block covers 16 batch rows x 64 hidden cols,
// each wave computes r/z/n gate tiles for the SAME 16 cols -> fused gates.
// h state kept double-buffered in fp32 (for the z*h blend) with a bf16
// shadow (WMMA A operand) -> LDS staging is a pure async byte copy.
// ---------------------------------------------------------------------------
__global__ __launch_bounds__(128) void gru_step(
    int t, const int* __restrict__ target,
    const __bf16* __restrict__ whhb, const float* __restrict__ b_hh,
    const float* __restrict__ gi,
    const float* __restrict__ h_cur, const __bf16* __restrict__ h_cur_bf,
    float* __restrict__ h_nxt, __bf16* __restrict__ h_nxt_bf,
    __bf16* __restrict__ h_allb)
{
  __shared__ __align__(32) __bf16 Hs[16 * HID];   // 32 KB
  const int rowTile = blockIdx.x;      // 0..1  (batch halves)
  const int colGrp  = blockIdx.y;      // 0..15
  const int tid  = threadIdx.x;
  const int wave = tid >> 5;
  const int lane = tid & 31;
  const int rowbase = rowTile * 16;

  // async global->LDS stage of the contiguous 16x1024 bf16 h tile (32 KB)
  {
    const unsigned ldsb = lds_addr_of(&Hs[0]);
    const char* src = (const char*)(h_cur_bf + (size_t)rowbase * HID);
    for (int ofs = tid * 16; ofs < 16 * HID * 2; ofs += 128 * 16)
      async_copy_b128(ldsb + ofs, src + ofs);
    wait_asynccnt0();
  }
  __syncthreads();

  const int nn = lane & 15, kh = lane >> 4;
  const int jbase = colGrp * 64 + wave * 16;
  const int col = jbase + nn;
  v8f aR = {}, aZ = {}, aN = {};
  const __bf16* ap = Hs + nn * HID + kh * 16;
  const __bf16* bR = whhb + (size_t)(          col) * HID + kh * 16;
  const __bf16* bZ = whhb + (size_t)(HID     + col) * HID + kh * 16;
  const __bf16* bN = whhb + (size_t)(2 * HID + col) * HID + kh * 16;
#pragma unroll 4
  for (int ks = 0; ks < HID / 32; ++ks) {
    v16bf a = *(const v16bf*)(ap + ks * 32);
    aR = wmma_bf16(a, *(const v16bf*)(bR + ks * 32), aR);
    aZ = wmma_bf16(a, *(const v16bf*)(bZ + ks * 32), aZ);
    aN = wmma_bf16(a, *(const v16bf*)(bN + ks * 32), aN);
  }

  const float bhr = b_hh[col], bhz = b_hh[HID + col], bhn = b_hh[2 * HID + col];
#pragma unroll
  for (int i = 0; i < 8; ++i) {
    int m = i + 8 * kh;
    int b = rowbase + m;               // batch index
    size_t girow = (size_t)t * BATCH + b;
    float ir  = gi[girow * G3 + col];
    float iz  = gi[girow * G3 + HID + col];
    float in_ = gi[girow * G3 + 2 * HID + col];
    float rg = 1.0f / (1.0f + __expf(-(ir + aR[i] + bhr)));
    float zg = 1.0f / (1.0f + __expf(-(iz + aZ[i] + bhz)));
    float ng = tanhf(in_ + rg * (aN[i] + bhn));
    float hp = h_cur[(size_t)b * HID + col];
    float hn = (1.0f - zg) * ng + zg * hp;
    h_allb[((size_t)t * BATCH + b) * HID + col] = (__bf16)hn;
    int tokp = (t == 0) ? SOS_ID : target[(t - 1) * BATCH + b];
    bool done = (t > 0) && (tokp == EOS_ID || tokp == PAD_ID);
    float hkeep = done ? hp : hn;
    h_nxt[(size_t)b * HID + col] = hkeep;
    h_nxt_bf[(size_t)b * HID + col] = (__bf16)hkeep;
  }
}

// ---------------------------------------------------------------------------
// Deferred output projection: logits = h_allb @ w_out^T + b_out.
// 2048 x 32000, K=1024.  WG: 256 thr (8 waves), tile 64 rows x 128 cols;
// A tile (128 KB bf16) staged via async global->LDS copy; w_out converted
// fp32->bf16 in flight (read exactly once), B frag reused 4x.
// ---------------------------------------------------------------------------
__global__ __launch_bounds__(256) void out_gemm(
    const __bf16* __restrict__ h_allb, const float* __restrict__ w_out,
    const float* __restrict__ b_out, float* __restrict__ out)
{
  __shared__ __align__(32) __bf16 As[64 * HID];   // 128 KB
  const int rowTile = blockIdx.x;      // 0..31
  const int colTile = blockIdx.y;      // 0..249
  const int tid  = threadIdx.x;
  const int wave = tid >> 5;
  const int lane = tid & 31;
  const int rowbase = rowTile * 64;

  // async global->LDS stage of the contiguous 64x1024 bf16 A tile (128 KB)
  {
    const unsigned ldsb = lds_addr_of(&As[0]);
    const char* src = (const char*)(h_allb + (size_t)rowbase * HID);
    for (int ofs = tid * 16; ofs < 64 * HID * 2; ofs += 256 * 16)
      async_copy_b128(ldsb + ofs, src + ofs);
    wait_asynccnt0();
  }
  __syncthreads();

  const int nn = lane & 15, kh = lane >> 4;
  const int colbase = colTile * 128 + wave * 16;
  const int col = colbase + nn;
  v8f acc0 = {}, acc1 = {}, acc2 = {}, acc3 = {};
  const float*  wrow = w_out + (size_t)col * HID + kh * 16;
  const __bf16* ap   = As + nn * HID + kh * 16;
#pragma unroll 2
  for (int ks = 0; ks < HID / 32; ++ks) {
    v16bf bfrag;
    const float4* wp = (const float4*)(wrow + ks * 32);
#pragma unroll
    for (int q = 0; q < 4; ++q) {
      float4 f = wp[q];
      bfrag[q * 4 + 0] = (__bf16)f.x;
      bfrag[q * 4 + 1] = (__bf16)f.y;
      bfrag[q * 4 + 2] = (__bf16)f.z;
      bfrag[q * 4 + 3] = (__bf16)f.w;
    }
    acc0 = wmma_bf16(*(const v16bf*)(ap + 0 * 16 * HID + ks * 32), bfrag, acc0);
    acc1 = wmma_bf16(*(const v16bf*)(ap + 1 * 16 * HID + ks * 32), bfrag, acc1);
    acc2 = wmma_bf16(*(const v16bf*)(ap + 2 * 16 * HID + ks * 32), bfrag, acc2);
    acc3 = wmma_bf16(*(const v16bf*)(ap + 3 * 16 * HID + ks * 32), bfrag, acc3);
  }
  float bias = b_out[col];
#pragma unroll
  for (int i = 0; i < 8; ++i) {
    int m = i + 8 * kh;
    out[(size_t)(rowbase +  0 + m) * VOCAB + col] = acc0[i] + bias;
    out[(size_t)(rowbase + 16 + m) * VOCAB + col] = acc1[i] + bias;
    out[(size_t)(rowbase + 32 + m) * VOCAB + col] = acc2[i] + bias;
    out[(size_t)(rowbase + 48 + m) * VOCAB + col] = acc3[i] + bias;
  }
}

// ---------------------------------------------------------------------------
// Row-wise softmax (in place on d_out) + done-masking to pad one-hot.
// One block per (t,b) row of 32000.
// ---------------------------------------------------------------------------
__global__ __launch_bounds__(256) void softmax_mask(
    const int* __restrict__ target, float* __restrict__ out)
{
  const int row = blockIdx.x;          // 0..2047 = t*BATCH + b
  const int t = row >> 5, b = row & 31;
  const int tid = threadIdx.x;
  float* p = out + (size_t)row * VOCAB;

  int tokp = (t == 0) ? SOS_ID : target[(t - 1) * BATCH + b];
  bool done = (t > 0) && (tokp == EOS_ID || tokp == PAD_ID);
  if (done) {                          // uniform per block
    for (int i = tid; i < VOCAB; i += 256) p[i] = (i == PAD_ID) ? 1.0f : 0.0f;
    return;
  }

  __shared__ float red[256];
  float mx = -3.4e38f;
  for (int i = tid; i < VOCAB; i += 256) mx = fmaxf(mx, p[i]);
  red[tid] = mx;
  __syncthreads();
  for (int s = 128; s > 0; s >>= 1) {
    if (tid < s) red[tid] = fmaxf(red[tid], red[tid + s]);
    __syncthreads();
  }
  mx = red[0];
  __syncthreads();

  float sum = 0.0f;
  for (int i = tid; i < VOCAB; i += 256) {
    float e = __expf(p[i] - mx);
    p[i] = e;
    sum += e;
  }
  red[tid] = sum;
  __syncthreads();
  for (int s = 128; s > 0; s >>= 1) {
    if (tid < s) red[tid] += red[tid + s];
    __syncthreads();
  }
  float inv = 1.0f / red[0];
  for (int i = tid; i < VOCAB; i += 256) p[i] *= inv;
}

// ---------------------------------------------------------------------------
extern "C" void kernel_launch(void* const* d_in, const int* in_sizes, int n_in,
                              void* d_out, int out_size, void* d_ws, size_t ws_size,
                              hipStream_t stream) {
  const float* hidden = (const float*)d_in[0];   // (1,32,1024)
  const int*   target = (const int*)  d_in[1];   // (64,32)
  // d_in[2] = lenseq (== SEQ)
  const float* emb    = (const float*)d_in[3];   // (32000,512)
  const float* w_ih   = (const float*)d_in[4];   // (3072,512)
  const float* w_hh   = (const float*)d_in[5];   // (3072,1024)
  const float* b_ih   = (const float*)d_in[6];   // (3072,)
  const float* b_hh   = (const float*)d_in[7];   // (3072,)
  const float* w_out  = (const float*)d_in[8];   // (32000,1024)
  const float* b_out  = (const float*)d_in[9];   // (32000,)
  float* out = (float*)d_out;                    // (64,32,32000)

  char* ws = (char*)d_ws;
  __bf16* wihb   = (__bf16*)ws;  ws += (size_t)G3 * EMB * sizeof(__bf16);
  __bf16* whhb   = (__bf16*)ws;  ws += (size_t)G3 * HID * sizeof(__bf16);
  float*  gi     = (float*)ws;   ws += (size_t)SEQ * BATCH * G3 * sizeof(float);
  __bf16* h_allb = (__bf16*)ws;  ws += (size_t)SEQ * BATCH * HID * sizeof(__bf16);
  float*  h0     = (float*)ws;   ws += (size_t)BATCH * HID * sizeof(float);
  float*  h1     = (float*)ws;   ws += (size_t)BATCH * HID * sizeof(float);
  __bf16* h0b    = (__bf16*)ws;  ws += (size_t)BATCH * HID * sizeof(__bf16);
  __bf16* h1b    = (__bf16*)ws;

  // 1) one-time weight casts to bf16 (+ bf16 shadow of initial h)
  {
    int n1 = G3 * EMB;
    cast_f32_bf16<<<(n1 + 255) / 256, 256, 0, stream>>>(w_ih, wihb, n1);
    int n2 = G3 * HID;
    cast_f32_bf16<<<(n2 + 255) / 256, 256, 0, stream>>>(w_hh, whhb, n2);
    int n3 = BATCH * HID;
    cast_f32_bf16<<<(n3 + 255) / 256, 256, 0, stream>>>(hidden, h0b, n3);
  }
  // 2) all-steps input-gate GEMM (tokens known ahead: teacher forcing)
  {
    dim3 grid(NROWS / 16, G3 / 128);
    gi_gemm<<<grid, 256, 0, stream>>>(target, emb, wihb, b_ih, gi);
  }
  // 3) serial GRU recurrence, double-buffered fp32 state + bf16 shadow
  hipMemcpyAsync(h0, hidden, (size_t)BATCH * HID * sizeof(float),
                 hipMemcpyDeviceToDevice, stream);
  float*  hc = h0;   float*  hn = h1;
  __bf16* hcb = h0b; __bf16* hnb = h1b;
  for (int t = 0; t < SEQ; ++t) {
    dim3 grid(2, HID / 64);
    gru_step<<<grid, 128, 0, stream>>>(t, target, whhb, b_hh, gi,
                                       hc, hcb, hn, hnb, h_allb);
    float*  tf = hc;  hc  = hn;  hn  = tf;
    __bf16* tb = hcb; hcb = hnb; hnb = tb;
  }
  // 4) deferred output projection (w_out read exactly once)
  {
    dim3 grid(NROWS / 64, VOCAB / 128);
    out_gemm<<<grid, 256, 0, stream>>>(h_allb, w_out, b_out, out);
  }
  // 5) softmax + pad-one-hot masking in place
  softmax_mask<<<NROWS, 256, 0, stream>>>(target, out);
}
